// Linear4Bit_21869973471373
// MI455X (gfx1250) — matrix operand order repaired and checked by reading, not compiled, not run on hardware
//
#include <hip/hip_runtime.h>

typedef _Float16 half2v __attribute__((ext_vector_type(2)));
typedef _Float16 half8  __attribute__((ext_vector_type(8)));
typedef _Float16 half16 __attribute__((ext_vector_type(16)));
typedef float    f32x8  __attribute__((ext_vector_type(8)));

#define K_TOTAL 4096
#define N_TOTAL 12288
#define BM 128
#define BN 128
#define BK 32
#define LDSPITCH 40            // BK + 8 halves of padding (80B row stride)
#define NKT (K_TOTAL / BK)     // 128 K-steps
#define SMEM_HALVES (2 * BM * LDSPITCH + 2 * BN * LDSPITCH)  // 20480 halves = 40KB

// C = x[M,K] * W[N,K]^T + bias, W dequantized from 4-bit on the fly.
// 8 waves (wave32) per block; block tile 128x128; wave tile 64x32 (4x2 WMMA frags).
// x tile: GLOBAL_LOAD_ASYNC_TO_LDS_B128 (ASYNCcnt).  W tile: int4 loads -> packed
// fp16 magic-number dequant (v_pk_fma_f16) -> ds_store_b128.
// Epilogue: per-wave LDS transpose bounce -> coalesced global_store_b128.
__global__ __launch_bounds__(256)
void linear4bit_wmma_kernel(const _Float16* __restrict__ x,
                            const int*      __restrict__ wq,     // one packed byte per int32
                            const _Float16* __restrict__ wnorm,  // [N]
                            const _Float16* __restrict__ bias,   // [N]
                            _Float16*       __restrict__ out)    // [M,N]
{
    __shared__ __align__(16) _Float16 smem[SMEM_HALVES];
    _Float16 (*Alds)[BM][LDSPITCH] = (_Float16(*)[BM][LDSPITCH])(smem);
    _Float16 (*Wlds)[BN][LDSPITCH] = (_Float16(*)[BN][LDSPITCH])(smem + 2 * BM * LDSPITCH);

    const int t    = threadIdx.x;
    const int lane = t & 31;
    const int wave = t >> 5;
    const int wm   = wave & 1;   // 0..1 -> M offset 0/64
    const int wn   = wave >> 1;  // 0..3 -> N offset 0/32/64/96
    const int m16  = lane & 15;
    const int hi   = lane >> 4;  // lane group (0: lanes 0-15, 1: lanes 16-31)

    const int m0 = blockIdx.y * BM;
    const int n0 = blockIdx.x * BN;

    // Loader mapping: 256 threads -> 128 rows x 2 16-half chunks of the 32-wide tile
    const int lrow = t >> 1;          // 0..127
    const int lcol = (t & 1) * 16;    // 0 or 16 (halves)

    // Per-output-row dequant constants (group == full row here).
    // w = q*s - n = (1024+q)*s + (-1024*s - n); (q|0x6400) as fp16 == 1024+q exactly.
    const float nf = (float)wnorm[n0 + lrow];
    const float sf = nf * (2.0f / 15.0f);
    const _Float16 sh = (_Float16)sf;
    const _Float16 bh = (_Float16)(-1024.0f * sf - nf);
    const half2v S2 = {sh, sh};
    const half2v B2 = {bh, bh};

    f32x8 acc[4][2] = {};

    const _Float16* aBase = x  + (size_t)(m0 + lrow) * K_TOTAL + lcol;
    const int*      qBase = wq + (size_t)(n0 + lrow) * (K_TOTAL / 2) + (t & 1) * 8;

    unsigned wst[8];  // 16 dequantized halves, packed 2 per dword in k-order

    // Async DMA: 32B of the x tile per thread straight into LDS (no VGPR staging).
    // The immediate offset is added to both the LDS and the global address.
    auto issueAsyncA = [&](int kt, int buf) {
        const _Float16* gp = aBase + (size_t)kt * BK;
        unsigned l0 = (unsigned)(uintptr_t)&Alds[buf][lrow][lcol];
        asm volatile("global_load_async_to_lds_b128 %0, %1, off\n\t"
                     "global_load_async_to_lds_b128 %0, %1, off offset:16"
                     :: "v"(l0), "v"(gp)
                     : "memory");
    };

    auto loadW = [&](int kt) {
        const int k0 = kt * BK;
        const int* qp = qBase + (k0 >> 1);
        int4 q0 = *(const int4*)(qp);
        int4 q1 = *(const int4*)(qp + 4);
        if (kt + 2 < NKT) {
            __builtin_prefetch(qBase + (((kt + 2) * BK) >> 1), 0, 1);
            __builtin_prefetch(aBase + (size_t)(kt + 2) * BK, 0, 1);
        }
        const unsigned qs[8] = {(unsigned)q0.x, (unsigned)q0.y, (unsigned)q0.z, (unsigned)q0.w,
                                (unsigned)q1.x, (unsigned)q1.y, (unsigned)q1.z, (unsigned)q1.w};
        #pragma unroll
        for (int e = 0; e < 8; ++e) {
            const unsigned b = qs[e];
            // low nibble -> fp16 lane 0 (k even), high nibble -> fp16 lane 1 (k odd)
            const unsigned p = ((b | (b << 12)) & 0x000F000Fu) | 0x64006400u;
            half2v h = __builtin_bit_cast(half2v, p);
            h = h * S2 + B2;                      // v_pk_fma_f16
            wst[e] = __builtin_bit_cast(unsigned, h);
        }
    };

    auto storeW = [&](int buf) {
        *(int4*)&Wlds[buf][lrow][lcol]     = make_int4((int)wst[0], (int)wst[1], (int)wst[2], (int)wst[3]);
        *(int4*)&Wlds[buf][lrow][lcol + 8] = make_int4((int)wst[4], (int)wst[5], (int)wst[6], (int)wst[7]);
    };

    auto compute = [&](int buf) {
        half16 afr[4];
        half16 bfr[2];
        // A fragment (16-bit 16x32 layout): elems 0..7  <- K = hi*8 + 0..7
        //                                   elems 8..15 <- K = 16 + hi*8 + 0..7
        #pragma unroll
        for (int i = 0; i < 4; ++i) {
            const _Float16* p = &Alds[buf][wm * 64 + i * 16 + m16][0];
            half8 x0 = *(const half8*)(p + hi * 8);
            half8 x1 = *(const half8*)(p + 16 + hi * 8);
            afr[i] = __builtin_shufflevector(x0, x1, 0,1,2,3,4,5,6,7,8,9,10,11,12,13,14,15);
        }
        // B fragment (col N = lane%16): elems 0..7 <- K = hi*8 + 0..7, elems 8..15 <- K = 16 + hi*8 + 0..7
        #pragma unroll
        for (int j = 0; j < 2; ++j) {
            const _Float16* p = &Wlds[buf][wn * 32 + j * 16 + m16][0];
            half8 y0 = *(const half8*)(p + hi * 8);
            half8 y1 = *(const half8*)(p + 16 + hi * 8);
            bfr[j] = __builtin_shufflevector(y0, y1, 0,1,2,3,4,5,6,7,8,9,10,11,12,13,14,15);
        }
        #pragma unroll
        for (int i = 0; i < 4; ++i)
            #pragma unroll
            for (int j = 0; j < 2; ++j)
                acc[i][j] = __builtin_amdgcn_wmma_f32_16x16x32_f16(
                    /*neg_a=*/false, afr[i], /*neg_b=*/false, bfr[j],
                    /*c_mod=*/(short)0, acc[i][j],
                    /*reuse_a=*/false, /*reuse_b=*/false);
    };

    // Prologue: stage K-step 0
    issueAsyncA(0, 0);
    loadW(0);
    storeW(0);
    asm volatile("s_wait_asynccnt 0x0" ::: "memory");
    __syncthreads();

    for (int kt = 0; kt < NKT; ++kt) {
        const int cur = kt & 1;
        const int nxt = cur ^ 1;
        if (kt + 1 < NKT) {
            issueAsyncA(kt + 1, nxt);  // nxt was freed by the barrier ending kt-1
            loadW(kt + 1);             // global int4 loads + packed fp16 dequant
        }
        compute(cur);                  // 8x v_wmma from LDS buffer `cur`
        if (kt + 1 < NKT) {
            storeW(nxt);                                       // ds_store_b128 x2
            asm volatile("s_wait_asynccnt 0x0" ::: "memory");  // A tile landed in LDS
            __syncthreads();                                   // all producers visible
        }
    }

    // ---- Epilogue: LDS-bounce transpose for coalesced b128 output stores ----
    __syncthreads();   // everyone done with the main-loop LDS before repurposing

    // Per-wave private stage region: 64 rows x LDSPITCH halves = 5KB (8 waves = 40KB)
    _Float16* stage = smem + wave * (64 * LDSPITCH);

    const _Float16 bv0 = bias[n0 + wn * 32 + m16];
    const _Float16 bv1 = bias[n0 + wn * 32 + 16 + m16];

    // Scatter C fragments (VGPR p -> M = p + hi*8, N = lane%16) into the stage tile
    #pragma unroll
    for (int i = 0; i < 4; ++i) {
        #pragma unroll
        for (int j = 0; j < 2; ++j) {
            const _Float16 bv = j ? bv1 : bv0;
            #pragma unroll
            for (int p = 0; p < 8; ++p) {
                stage[(i * 16 + hi * 8 + p) * LDSPITCH + j * 16 + m16] =
                    (_Float16)acc[i][j][p] + bv;
            }
        }
    }

    // Same-wave LDS ops are in-order and the region is wave-private: no barrier.
    // Read back row-major: 8 rows per instruction (4 lanes x 16B cover one 64B row).
    #pragma unroll
    for (int g = 0; g < 8; ++g) {
        const int r2 = g * 8 + (lane >> 2);     // 0..63
        const int c2 = (lane & 3) * 8;          // 0/8/16/24
        half8 v = *(const half8*)&stage[r2 * LDSPITCH + c2];
        *(half8*)&out[(size_t)(m0 + wm * 64 + r2) * N_TOTAL + (n0 + wn * 32 + c2)] = v;
    }
}

extern "C" void kernel_launch(void* const* d_in, const int* in_sizes, int n_in,
                              void* d_out, int out_size, void* d_ws, size_t ws_size,
                              hipStream_t stream) {
    const _Float16* x   = (const _Float16*)d_in[0];  // [8,2048,4096] fp16
    const int*      wq  = (const int*)d_in[1];       // [12288,2048] int32 (one byte each)
    const _Float16* wn  = (const _Float16*)d_in[2];  // [12288,1] fp16
    const _Float16* bs  = (const _Float16*)d_in[3];  // [12288] fp16
    _Float16*       out = (_Float16*)d_out;          // [8,2048,12288] fp16

    const int M = in_sizes[0] / K_TOTAL;             // 16384
    dim3 grid(N_TOTAL / BN, M / BM);                 // (96, 128)
    linear4bit_wmma_kernel<<<grid, dim3(256), 0, stream>>>(x, wq, wn, bs, out);
}
